// CGCNN_7275674599846
// MI455X (gfx1250) — matrix operand order repaired and checked by reading, not compiled, not run on hardware
//
#include <hip/hip_runtime.h>
#include <hip/hip_bf16.h>

// ---------------------------------------------------------------------------
// CGCNN forward for MI455X (gfx1250): bf16 WMMA for all GEMMs, fused
// gather->WMMA edge MLP, atomic scatter-add aggregation.
// M=32 per wave in the heavy kernels: each B (weight) fragment feeds two
// WMMAs, halving L2 weight traffic and doubling matrix-op density.
// ---------------------------------------------------------------------------

typedef __attribute__((ext_vector_type(16))) __bf16 v16bf;
typedef __attribute__((ext_vector_type(8)))  float  v8f;

union AF {
    v16bf v;
    unsigned int u[8];
};

// ISA 16-bit A-matrix 16x32 layout: dword j of lane (sub = lane>>4) holds
// K = kpair(j,sub), kpair(j,sub)+1  for row M = lane&15.
__device__ __forceinline__ int kpair(int j, int sub) {
    return (j < 4) ? (2 * j + 8 * sub) : (16 + 2 * (j - 4) + 8 * sub);
}

__device__ __forceinline__ v8f wmma_bf16(v16bf a, v16bf b, v8f c) {
    return __builtin_amdgcn_wmma_f32_16x16x32_bf16(false, a, false, b,
                                                   (short)0, c, false, false);
}

__device__ __forceinline__ unsigned short f2bfu(float f) {
    return __builtin_bit_cast(unsigned short, (__bf16)f);
}
__device__ __forceinline__ unsigned int pack2(float a, float b) {
    return (unsigned int)f2bfu(a) | ((unsigned int)f2bfu(b) << 16);
}

__device__ __forceinline__ float softplus_f(float x) {
    return (x > 20.f) ? x : log1pf(__expf(x));
}

// B-fragment (32x16, K x N) from transposed bf16 weights Wt[outCh][Krow]:
// dword j holds K = kbase + kpair(j,sub) for column n = lane&15.
__device__ __forceinline__ v16bf load_bfrag(const unsigned short* __restrict__ Wt,
                                            int Krow, int colbase, int n,
                                            int kbase, int sub) {
    AF b;
    const unsigned short* p = Wt + (size_t)(colbase + n) * Krow + kbase;
#pragma unroll
    for (int j = 0; j < 8; ++j)
        b.u[j] = *(const unsigned int*)(p + kpair(j, sub));
    return b.v;
}

// A-fragment from a contiguous bf16 row (global or LDS), chunk start = src.
__device__ __forceinline__ v16bf load_afrag_bf(const __bf16* src, int sub) {
    AF a;
#pragma unroll
    for (int j = 0; j < 8; ++j)
        a.u[j] = *(const unsigned int*)(src + kpair(j, sub));
    return a.v;
}

// A-fragment from a contiguous fp32 row (convert+pack inline).
__device__ __forceinline__ v16bf load_afrag_f32(const float* src, int sub) {
    AF a;
#pragma unroll
    for (int j = 0; j < 8; ++j) {
        int k = kpair(j, sub);
        a.u[j] = pack2(src[k], src[k + 1]);
    }
    return a.v;
}

__device__ __forceinline__ void atomicMaxF(float* addr, float val) {
    unsigned int* ua = (unsigned int*)addr;
    unsigned int old = *ua;
    while (__uint_as_float(old) < val) {
        unsigned int assumed = old;
        old = atomicCAS(ua, assumed, __float_as_uint(val));
        if (old == assumed) break;
    }
}

#define HDIM 128
#define LDS_STRIDE 136   // 16x128 bf16 tile padded to avoid bank conflicts

// ---------------------------------------------------------------------------
// RBF expansion: rbf_bf16[e][i] = exp(-0.5*((d-c_i)/0.5)^2) * cosine-cutoff
// ---------------------------------------------------------------------------
__global__ void rbf_kernel(const float* __restrict__ d, __bf16* __restrict__ out, int E) {
    int idx = blockIdx.x * blockDim.x + threadIdx.x;
    if (idx >= E * 64) return;
    int e = idx >> 6, i = idx & 63;
    float x = d[e];
    float cut = (x < 8.f) ? 0.5f * (__cosf(3.14159265358979f * x * 0.125f) + 1.f) : 0.f;
    float c = (8.f / 63.f) * (float)i;
    float t = (x - c) * 2.0f;           // / WIDTH (0.5)
    out[idx] = (__bf16)(__expf(-0.5f * t * t) * cut);
}

// ---------------------------------------------------------------------------
// fp32 [K][N] weights -> bf16 transposed [N][Kpad] (zero pad K >= Ksrc)
// ---------------------------------------------------------------------------
__global__ void conv_wT(const float* __restrict__ src, unsigned short* __restrict__ dst,
                        int Ksrc, int Kpad, int Nc) {
    int idx = blockIdx.x * blockDim.x + threadIdx.x;
    if (idx >= Nc * Kpad) return;
    int n = idx / Kpad, k = idx % Kpad;
    float v = (k < Ksrc) ? src[(size_t)k * Nc + n] : 0.f;
    dst[idx] = f2bfu(v);
}

// ---------------------------------------------------------------------------
// Embedding: h[N,128] = X[N,92] @ W_emb + b (WMMA, K padded to 96)
// ---------------------------------------------------------------------------
__global__ void emb_gemm(const float* __restrict__ X,
                         const unsigned short* __restrict__ WT,   // [128][96] bf16
                         const float* __restrict__ bias,
                         float* __restrict__ h, __bf16* __restrict__ hbf, int N) {
    int wave = (int)((blockIdx.x * blockDim.x + threadIdx.x) >> 5);
    int lane = threadIdx.x & 31;
    int m0 = wave * 16;
    if (m0 >= N) return;
    int sub = lane >> 4, n = lane & 15, mrow = lane & 15;
    int mr = m0 + mrow; if (mr > N - 1) mr = N - 1;
    const float* xr = X + (size_t)mr * 92;

    v8f acc[8] = {};
#pragma unroll
    for (int kt = 0; kt < 3; ++kt) {
        int kbase = 32 * kt;
        AF a;
#pragma unroll
        for (int j = 0; j < 8; ++j) {
            int k = kbase + kpair(j, sub);
            float f0 = (k < 92) ? xr[k] : 0.f;
            float f1 = (k + 1 < 92) ? xr[k + 1] : 0.f;
            a.u[j] = pack2(f0, f1);
        }
#pragma unroll
        for (int nt = 0; nt < 8; ++nt)
            acc[nt] = wmma_bf16(a.v, load_bfrag(WT, 96, nt * 16, n, kbase, sub), acc[nt]);
    }
#pragma unroll
    for (int r = 0; r < 8; ++r) {
        int m = m0 + 8 * sub + r;
        if (m < N) {
#pragma unroll
            for (int nt = 0; nt < 8; ++nt) {
                int c = nt * 16 + n;
                float v = acc[nt][r] + bias[c];
                h[(size_t)m * HDIM + c] = v;
                hbf[(size_t)m * HDIM + c] = (__bf16)v;
            }
        }
    }
}

// ---------------------------------------------------------------------------
// Edge message MLP (fused), 32 edges per wave:
//   hidden = softplus([h[row]|h[col]|rbf] @ W1 + b1)   (K = 320)
//   msg    = hidden @ W2 + b2                           (K = 128)
//   agg[row] += msg   (f32 atomics)
// ---------------------------------------------------------------------------
__global__ void edge_msg(const __bf16* __restrict__ hbf, const __bf16* __restrict__ rbf,
                         const unsigned short* __restrict__ W1T,  // [128][320]
                         const float* __restrict__ b1,
                         const unsigned short* __restrict__ W2T,  // [128][128]
                         const float* __restrict__ b2,
                         const int* __restrict__ rowi, const int* __restrict__ coli,
                         float* __restrict__ agg, int E) {
    __shared__ __bf16 lds[4][2][16 * LDS_STRIDE];
    int wid  = threadIdx.x >> 5;
    int wave = (int)((blockIdx.x * blockDim.x + threadIdx.x) >> 5);
    int lane = threadIdx.x & 31;
    int e0 = wave * 32;
    bool valid = (e0 + 32 <= E);
    int sub = lane >> 4, n = lane & 15, mrow = lane & 15;

    int ea = e0 + mrow;      if (ea > E - 1) ea = E - 1;
    int eb = e0 + 16 + mrow; if (eb > E - 1) eb = E - 1;
    const __bf16* rpa = hbf + (size_t)rowi[ea] * HDIM;
    const __bf16* cpa = hbf + (size_t)coli[ea] * HDIM;
    const __bf16* gpa = rbf + (size_t)ea * 64;
    const __bf16* rpb = hbf + (size_t)rowi[eb] * HDIM;
    const __bf16* cpb = hbf + (size_t)coli[eb] * HDIM;
    const __bf16* gpb = rbf + (size_t)eb * 64;
    __builtin_prefetch(cpa, 0, 0);   // global_prefetch_b8: warm col gathers
    __builtin_prefetch(cpb, 0, 0);

    // ---- GEMM1: 32x320 @ 320x128 ----
    v8f acc[2][8] = {};
#pragma unroll
    for (int kt = 0; kt < 10; ++kt) {
        const __bf16 *sa, *sb; int off;
        if (kt < 4)      { sa = rpa; sb = rpb; off = 32 * kt; }
        else if (kt < 8) { sa = cpa; sb = cpb; off = 32 * kt - 128; }
        else             { sa = gpa; sb = gpb; off = 32 * kt - 256; }
        v16bf a0 = load_afrag_bf(sa + off, sub);
        v16bf a1 = load_afrag_bf(sb + off, sub);
#pragma unroll
        for (int nt = 0; nt < 8; ++nt) {
            v16bf b = load_bfrag(W1T, 320, nt * 16, n, 32 * kt, sub);
            acc[0][nt] = wmma_bf16(a0, b, acc[0][nt]);
            acc[1][nt] = wmma_bf16(a1, b, acc[1][nt]);
        }
    }

    // bias + softplus, stash hidden tiles to LDS (C-layout -> row-major bf16)
#pragma unroll
    for (int t = 0; t < 2; ++t) {
        __bf16* myl = &lds[wid][t][0];
#pragma unroll
        for (int r = 0; r < 8; ++r) {
            int m = 8 * sub + r;
#pragma unroll
            for (int nt = 0; nt < 8; ++nt) {
                int c = nt * 16 + n;
                myl[m * LDS_STRIDE + c] = (__bf16)softplus_f(acc[t][nt][r] + b1[c]);
            }
        }
    }
    __syncthreads();

    // ---- GEMM2: 32x128 @ 128x128 ----
    v8f acc2[2][8] = {};
#pragma unroll
    for (int kt = 0; kt < 4; ++kt) {
        v16bf a0 = load_afrag_bf(&lds[wid][0][mrow * LDS_STRIDE + 32 * kt], sub);
        v16bf a1 = load_afrag_bf(&lds[wid][1][mrow * LDS_STRIDE + 32 * kt], sub);
#pragma unroll
        for (int nt = 0; nt < 8; ++nt) {
            v16bf b = load_bfrag(W2T, 128, nt * 16, n, 32 * kt, sub);
            acc2[0][nt] = wmma_bf16(a0, b, acc2[0][nt]);
            acc2[1][nt] = wmma_bf16(a1, b, acc2[1][nt]);
        }
    }

    // scatter-add into agg[row]
    if (valid) {
#pragma unroll
        for (int t = 0; t < 2; ++t) {
#pragma unroll
            for (int r = 0; r < 8; ++r) {
                int em = e0 + 16 * t + 8 * sub + r;
                float* ap = agg + (size_t)rowi[em] * HDIM;
#pragma unroll
                for (int nt = 0; nt < 8; ++nt) {
                    int c = nt * 16 + n;
                    atomicAdd(ap + c, acc2[t][nt][r] + b2[c]);
                }
            }
        }
    }
}

// ---------------------------------------------------------------------------
// Node update MLP, 32 nodes per wave:
//   u = softplus([h|agg] @ U1 + b1) @ U2 + b2   (K=256 then 128)
// ---------------------------------------------------------------------------
__global__ void node_upd(const __bf16* __restrict__ hbf, const float* __restrict__ agg,
                         const unsigned short* __restrict__ W1T,  // [128][256]
                         const float* __restrict__ b1,
                         const unsigned short* __restrict__ W2T,  // [128][128]
                         const float* __restrict__ b2,
                         float* __restrict__ u, int N) {
    __shared__ __bf16 lds[4][2][16 * LDS_STRIDE];
    int wid  = threadIdx.x >> 5;
    int wave = (int)((blockIdx.x * blockDim.x + threadIdx.x) >> 5);
    int lane = threadIdx.x & 31;
    int n0 = wave * 32;
    bool valid = (n0 + 32 <= N);
    int sub = lane >> 4, n = lane & 15, mrow = lane & 15;

    int na = n0 + mrow;      if (na > N - 1) na = N - 1;
    int nb = n0 + 16 + mrow; if (nb > N - 1) nb = N - 1;
    const __bf16* hpa = hbf + (size_t)na * HDIM;
    const float*  apa = agg + (size_t)na * HDIM;
    const __bf16* hpb = hbf + (size_t)nb * HDIM;
    const float*  apb = agg + (size_t)nb * HDIM;

    v8f acc[2][8] = {};
#pragma unroll
    for (int kt = 0; kt < 8; ++kt) {
        v16bf a0, a1;
        if (kt < 4) {
            a0 = load_afrag_bf(hpa + 32 * kt, sub);
            a1 = load_afrag_bf(hpb + 32 * kt, sub);
        } else {
            int off = 32 * kt - 128;
            a0 = load_afrag_f32(apa + off, sub);
            a1 = load_afrag_f32(apb + off, sub);
        }
#pragma unroll
        for (int nt = 0; nt < 8; ++nt) {
            v16bf b = load_bfrag(W1T, 256, nt * 16, n, 32 * kt, sub);
            acc[0][nt] = wmma_bf16(a0, b, acc[0][nt]);
            acc[1][nt] = wmma_bf16(a1, b, acc[1][nt]);
        }
    }

#pragma unroll
    for (int t = 0; t < 2; ++t) {
        __bf16* myl = &lds[wid][t][0];
#pragma unroll
        for (int r = 0; r < 8; ++r) {
            int m = 8 * sub + r;
#pragma unroll
            for (int nt = 0; nt < 8; ++nt) {
                int c = nt * 16 + n;
                myl[m * LDS_STRIDE + c] = (__bf16)softplus_f(acc[t][nt][r] + b1[c]);
            }
        }
    }
    __syncthreads();

    v8f acc2[2][8] = {};
#pragma unroll
    for (int kt = 0; kt < 4; ++kt) {
        v16bf a0 = load_afrag_bf(&lds[wid][0][mrow * LDS_STRIDE + 32 * kt], sub);
        v16bf a1 = load_afrag_bf(&lds[wid][1][mrow * LDS_STRIDE + 32 * kt], sub);
#pragma unroll
        for (int nt = 0; nt < 8; ++nt) {
            v16bf b = load_bfrag(W2T, 128, nt * 16, n, 32 * kt, sub);
            acc2[0][nt] = wmma_bf16(a0, b, acc2[0][nt]);
            acc2[1][nt] = wmma_bf16(a1, b, acc2[1][nt]);
        }
    }

    if (valid) {
#pragma unroll
        for (int t = 0; t < 2; ++t) {
#pragma unroll
            for (int r = 0; r < 8; ++r) {
                int m = n0 + 16 * t + 8 * sub + r;
#pragma unroll
                for (int nt = 0; nt < 8; ++nt) {
                    int c = nt * 16 + n;
                    u[(size_t)m * HDIM + c] = acc2[t][nt][r] + b2[c];
                }
            }
        }
    }
}

// ---------------------------------------------------------------------------
// BatchNorm (training stats) + residual into h / hbf
// ---------------------------------------------------------------------------
__global__ void bn_stats(const float* __restrict__ u, float* __restrict__ stats, int N) {
    int c = threadIdx.x;   // 128 threads
    float s = 0.f, s2 = 0.f;
    for (int n = blockIdx.x; n < N; n += gridDim.x) {
        float v = u[(size_t)n * HDIM + c];
        s += v; s2 += v * v;
    }
    atomicAdd(&stats[c], s);
    atomicAdd(&stats[HDIM + c], s2);
}

__global__ void bn_apply(float* __restrict__ h, __bf16* __restrict__ hbf,
                         const float* __restrict__ u, const float* __restrict__ stats,
                         const float* __restrict__ gamma, const float* __restrict__ beta,
                         int N) {
    int idx = blockIdx.x * blockDim.x + threadIdx.x;
    if (idx >= N * HDIM) return;
    int c = idx & (HDIM - 1);
    float invN = 1.f / (float)N;
    float mu = stats[c] * invN;
    float var = stats[HDIM + c] * invN - mu * mu;
    float xn = (u[idx] - mu) * rsqrtf(var + 1e-5f) * gamma[c] + beta[c];
    float nh = h[idx] + xn;
    h[idx] = nh;
    hbf[idx] = (__bf16)nh;
}

// ---------------------------------------------------------------------------
// Pooling + readout
// ---------------------------------------------------------------------------
__global__ void pool_init(float* hsum, float* hmax, float* cnt) {
    int i = blockIdx.x * blockDim.x + threadIdx.x;
    if (i < 64 * HDIM) { hsum[i] = 0.f; hmax[i] = -3.402823466e38f; }
    if (i < 64) cnt[i] = 0.f;
}

__global__ void pool_kernel(const float* __restrict__ h, const int* __restrict__ batch,
                            float* __restrict__ hsum, float* __restrict__ hmax,
                            float* __restrict__ cnt, int N) {
    int idx = blockIdx.x * blockDim.x + threadIdx.x;
    if (idx >= N * HDIM) return;
    int nd = idx >> 7, c = idx & (HDIM - 1);
    int g = batch[nd];
    float v = h[idx];
    atomicAdd(&hsum[g * HDIM + c], v);
    atomicMaxF(&hmax[g * HDIM + c], v);
    if (c == 0) atomicAdd(&cnt[g], 1.f);
}

__global__ void readout(const float* __restrict__ hsum, const float* __restrict__ hmax,
                        const float* __restrict__ cnt,
                        const float* __restrict__ W1, const float* __restrict__ b1,
                        const float* __restrict__ W2, const float* __restrict__ b2,
                        const float* __restrict__ oW, const float* __restrict__ ob,
                        float* __restrict__ out) {
    __shared__ float pooled[2 * HDIM];
    __shared__ float t1[HDIM];
    __shared__ float red[HDIM];
    int g = blockIdx.x, c = threadIdx.x;   // 128 threads
    float ct = fmaxf(cnt[g], 1.f);
    pooled[c] = hsum[g * HDIM + c] / ct;
    pooled[HDIM + c] = hmax[g * HDIM + c];
    __syncthreads();
    float a = b1[c];
    for (int k = 0; k < 2 * HDIM; ++k) a += pooled[k] * W1[k * HDIM + c];
    t1[c] = softplus_f(a);
    __syncthreads();
    float t = b2[c];
    for (int k = 0; k < HDIM; ++k) t += t1[k] * W2[k * HDIM + c];
    red[c] = t * oW[c];
    __syncthreads();
    for (int s = 64; s > 0; s >>= 1) {
        if (c < s) red[c] += red[c + s];
        __syncthreads();
    }
    if (c == 0) out[g] = red[0] + ob[0];
}

// ---------------------------------------------------------------------------
// Host launcher
// ---------------------------------------------------------------------------
extern "C" void kernel_launch(void* const* d_in, const int* in_sizes, int n_in,
                              void* d_out, int out_size, void* d_ws, size_t ws_size,
                              hipStream_t stream) {
    const float* nodef = (const float*)d_in[0];
    const int*   eidx  = (const int*)d_in[1];
    const float* eattr = (const float*)d_in[2];
    const int*   batch = (const int*)d_in[3];
    const float* W_emb = (const float*)d_in[4];
    const float* b_emb = (const float*)d_in[5];
    const float* msgW1 = (const float*)d_in[6];
    const float* msgb1 = (const float*)d_in[7];
    const float* msgW2 = (const float*)d_in[8];
    const float* msgb2 = (const float*)d_in[9];
    const float* updW1 = (const float*)d_in[10];
    const float* updb1 = (const float*)d_in[11];
    const float* updW2 = (const float*)d_in[12];
    const float* updb2 = (const float*)d_in[13];
    const float* bng   = (const float*)d_in[14];
    const float* bnb   = (const float*)d_in[15];
    const float* fcW1  = (const float*)d_in[16];
    const float* fcb1  = (const float*)d_in[17];
    const float* fcW2  = (const float*)d_in[18];
    const float* fcb2  = (const float*)d_in[19];
    const float* outW  = (const float*)d_in[20];
    const float* outb  = (const float*)d_in[21];
    float* out = (float*)d_out;

    const int N = in_sizes[0] / 92;
    const int E = in_sizes[2];
    const int L = 3;
    const int*  rowi = eidx;
    const int*  coli = eidx + E;

    // -------- workspace carve-out --------
    char* w = (char*)d_ws;
    auto alloc = [&](size_t bytes) {
        void* p = (void*)w;
        w += (bytes + 255) & ~(size_t)255;
        return p;
    };
    float*  h    = (float*)alloc((size_t)N * HDIM * 4);
    float*  u    = (float*)alloc((size_t)N * HDIM * 4);
    float*  agg  = (float*)alloc((size_t)N * HDIM * 4);
    __bf16* hbf  = (__bf16*)alloc((size_t)N * HDIM * 2);
    __bf16* rbf  = (__bf16*)alloc((size_t)E * 64 * 2);
    unsigned short* WembT  = (unsigned short*)alloc((size_t)128 * 96 * 2);
    unsigned short* msgW1T = (unsigned short*)alloc((size_t)L * 128 * 320 * 2);
    unsigned short* msgW2T = (unsigned short*)alloc((size_t)L * 128 * 128 * 2);
    unsigned short* updW1T = (unsigned short*)alloc((size_t)L * 128 * 256 * 2);
    unsigned short* updW2T = (unsigned short*)alloc((size_t)L * 128 * 128 * 2);
    float* stats = (float*)alloc(256 * 4);
    float* hsum  = (float*)alloc(64 * HDIM * 4);
    float* hmax  = (float*)alloc(64 * HDIM * 4);
    float* cnt   = (float*)alloc(64 * 4);

    // -------- weight conversion (fp32 [K][N] -> bf16 transposed [N][Kpad]) ----
    conv_wT<<<(128 * 96 + 255) / 256, 256, 0, stream>>>(W_emb, WembT, 92, 96, 128);
    for (int l = 0; l < L; ++l) {
        conv_wT<<<(128 * 320 + 255) / 256, 256, 0, stream>>>(
            msgW1 + (size_t)l * 320 * 128, msgW1T + (size_t)l * 128 * 320, 320, 320, 128);
        conv_wT<<<(128 * 128 + 255) / 256, 256, 0, stream>>>(
            msgW2 + (size_t)l * 128 * 128, msgW2T + (size_t)l * 128 * 128, 128, 128, 128);
        conv_wT<<<(128 * 256 + 255) / 256, 256, 0, stream>>>(
            updW1 + (size_t)l * 256 * 128, updW1T + (size_t)l * 128 * 256, 256, 256, 128);
        conv_wT<<<(128 * 128 + 255) / 256, 256, 0, stream>>>(
            updW2 + (size_t)l * 128 * 128, updW2T + (size_t)l * 128 * 128, 128, 128, 128);
    }

    // -------- RBF + embedding --------
    rbf_kernel<<<((size_t)E * 64 + 255) / 256, 256, 0, stream>>>(eattr, rbf, E);
    {
        int tiles = (N + 15) / 16;
        int blocks = (tiles + 3) / 4;
        emb_gemm<<<blocks, 128, 0, stream>>>(nodef, WembT, b_emb, h, hbf, N);
    }

    // -------- conv layers --------
    int ejobs   = (E + 31) / 32;           // 32 edges per wave
    int eblocks = (ejobs + 3) / 4;         // 4 waves per block
    int njobs   = (N + 31) / 32;
    int nblocks = (njobs + 3) / 4;
    for (int l = 0; l < L; ++l) {
        hipMemsetAsync(agg, 0, (size_t)N * HDIM * 4, stream);
        edge_msg<<<eblocks, 128, 0, stream>>>(
            hbf, rbf,
            msgW1T + (size_t)l * 128 * 320, msgb1 + l * HDIM,
            msgW2T + (size_t)l * 128 * 128, msgb2 + l * HDIM,
            rowi, coli, agg, E);
        node_upd<<<nblocks, 128, 0, stream>>>(
            hbf, agg,
            updW1T + (size_t)l * 128 * 256, updb1 + l * HDIM,
            updW2T + (size_t)l * 128 * 128, updb2 + l * HDIM,
            u, N);
        hipMemsetAsync(stats, 0, 256 * 4, stream);
        bn_stats<<<512, 128, 0, stream>>>(u, stats, N);
        bn_apply<<<((size_t)N * HDIM + 255) / 256, 256, 0, stream>>>(
            h, hbf, u, stats, bng + l * HDIM, bnb + l * HDIM, N);
    }

    // -------- pooling + readout --------
    pool_init<<<(64 * HDIM + 255) / 256, 256, 0, stream>>>(hsum, hmax, cnt);
    pool_kernel<<<((size_t)N * HDIM + 255) / 256, 256, 0, stream>>>(h, batch, hsum, hmax, cnt, N);
    readout<<<64, 128, 0, stream>>>(hsum, hmax, cnt, fcW1, fcb1, fcW2, fcb2, outW, outb, out);
}
